// DynamicVoxelEncoder_54855322304605
// MI455X (gfx1250) — matrix operand order repaired and checked by reading, not compiled, not run on hardware
//
#include <hip/hip_runtime.h>
#include <hip/hip_bf16.h>
#include <math.h>

#define NPTS      400000
#define CANVAS_SZ 281600
#define CXDIM     352
#define PLANE     140800   /* CY*CX = 400*352 */

typedef __attribute__((ext_vector_type(16))) __bf16        v16bf;
typedef __attribute__((ext_vector_type(8)))  float         v8f;
typedef __attribute__((ext_vector_type(8)))  unsigned int  v8u;

// Truncation split: hi = f with low 16 mantissa bits cleared (exact: lo = f-hi),
// both halves packed into one dword via v_perm_b32. Element 2v -> low 16 bits.
static __device__ __forceinline__ void splitpack(float f0, float f1,
                                                 unsigned& hi, unsigned& lo) {
    unsigned u0 = __float_as_uint(f0), u1 = __float_as_uint(f1);
    float l0 = f0 - __uint_as_float(u0 & 0xFFFF0000u);
    float l1 = f1 - __uint_as_float(u1 & 0xFFFF0000u);
    hi = __builtin_amdgcn_perm(u1, u0, 0x07060302u);
    lo = __builtin_amdgcn_perm(__float_as_uint(l1), __float_as_uint(l0), 0x07060302u);
}

static __device__ __forceinline__ v8f wmma_bf16(v16bf a, v16bf b, v8f c) {
    return __builtin_amdgcn_wmma_f32_16x16x32_bf16(false, a, false, b, (short)0, c,
                                                   false, false);
}

// ---------------------------------------------------------------- zero
__global__ void zero_kernel(float4* __restrict__ p, long n4) {
    for (long i = (long)blockIdx.x * blockDim.x + threadIdx.x; i < n4;
         i += (long)gridDim.x * blockDim.x)
        p[i] = make_float4(0.f, 0.f, 0.f, 0.f);
}

// ---------------- pre-pack a weight matrix into WMMA B-fragment layout -------
// out[((ks*ntiles + nt)*32 + lane)*16 + {0..7 hi, 8..15 lo}]
__global__ void packW_kernel(const float* __restrict__ W, unsigned* __restrict__ out,
                             int ncols, int ntiles, int ksteps, int kvalid) {
    int idx = blockIdx.x * blockDim.x + threadIdx.x;
    if (idx >= ksteps * ntiles * 32) return;
    int lane = idx & 31;
    int nt   = (idx >> 5) % ntiles;
    int ks   = (idx >> 5) / ntiles;
    int half = lane >> 4, ncol = lane & 15;
    int n  = nt * 16 + ncol;
    int kb = ks * 32;
    unsigned* o = out + (size_t)idx * 16;
#pragma unroll
    for (int v = 0; v < 8; ++v) {
        int k = kb + 16 * half + 2 * v;
        float f0 = (k     < kvalid) ? W[k * ncols + n]       : 0.f;
        float f1 = (k + 1 < kvalid) ? W[(k + 1) * ncols + n] : 0.f;
        unsigned h, l;
        splitpack(f0, f1, h, l);
        o[v]     = h;
        o[8 + v] = l;
    }
}

// ------------------------------------------------- per-point voxel stats
__global__ void point_stats_kernel(const float* __restrict__ feat,
                                   const int* __restrict__ coors,
                                   int* __restrict__ lin,
                                   float* __restrict__ stats) {
    int i = blockIdx.x * blockDim.x + threadIdx.x;
    if (i >= NPTS) return;
    int b = coors[4*i+0], z = coors[4*i+1], y = coors[4*i+2], x = coors[4*i+3];
    int l = (b + z) * PLANE + y * CXDIM + x;       // CZ==1
    lin[i] = l;
    atomicAdd(&stats[4*l+0], 1.0f);
    atomicAdd(&stats[4*l+1], feat[15*i+0]);
    atomicAdd(&stats[4*l+2], feat[15*i+1]);
    atomicAdd(&stats[4*l+3], feat[15*i+2]);
}

// ----------------------------------------------- build padded 22->32 feats
__global__ void build_x22_kernel(const float* __restrict__ feat,
                                 const int* __restrict__ coors,
                                 const int* __restrict__ lin,
                                 const float* __restrict__ stats,
                                 float* __restrict__ x22) {
    int i = blockIdx.x * blockDim.x + threadIdx.x;
    if (i >= NPTS) return;
    const float* f = feat + 15 * i;
    int l = lin[i];
    float inv = 1.0f / stats[4*l+0];
    float mx = stats[4*l+1] * inv, my = stats[4*l+2] * inv, mz = stats[4*l+3] * inv;
    float cz = (float)coors[4*i+1], cy = (float)coors[4*i+2], cx = (float)coors[4*i+3];
    float* o = x22 + (size_t)i * 32;
    o[0]=f[0]; o[1]=f[1]; o[2]=f[2]; o[3]=f[3]; o[4]=f[4];
    o[5]=f[0]-mx; o[6]=f[1]-my; o[7]=f[2]-mz;
    o[8]  = f[0] - (cx * 0.2f + 0.1f);
    o[9]  = f[1] - (cy * 0.2f - 39.9f);
    o[10] = f[2] - (cz * 4.0f - 1.0f);
    o[11] = sqrtf(f[0]*f[0] + f[1]*f[1] + f[2]*f[2]);
#pragma unroll
    for (int j = 0; j < 10; ++j) o[12+j] = f[5+j];
#pragma unroll
    for (int j = 22; j < 32; ++j) o[j] = 0.f;
}

// ------------------------------------------- GEMM0: y0 = x22[ N x 32 ] * W0
__global__ __launch_bounds__(128)
void gemm0_kernel(const float* __restrict__ x22, const unsigned* __restrict__ wpack,
                  float* __restrict__ y0, float* __restrict__ bnSum,
                  float* __restrict__ bnSq) {
    int tid  = threadIdx.x;
    int wave = tid >> 5, lane = tid & 31;
    int half = lane >> 4, mrow = lane & 15, ncol = lane & 15;
    int row0 = (blockIdx.x * 4 + wave) * 16;

    // A fragment (16x32 bf16, hi/lo split), packed via v_perm
    v8u ah, al;
    const float* ar = x22 + (size_t)(row0 + mrow) * 32;
#pragma unroll
    for (int v = 0; v < 8; ++v) {
        int k = 2*v + 16*half;
        unsigned h, l;
        splitpack(ar[k], ar[k+1], h, l);
        ah[v] = h; al[v] = l;
    }
    v16bf ahi = __builtin_bit_cast(v16bf, ah);
    v16bf alo = __builtin_bit_cast(v16bf, al);

#pragma unroll
    for (int nt = 0; nt < 4; ++nt) {
        int n = nt*16 + ncol;
        const v8u* wp = (const v8u*)(wpack + (size_t)(nt * 32 + lane) * 16);
        v16bf bhi = __builtin_bit_cast(v16bf, wp[0]);
        v16bf blo = __builtin_bit_cast(v16bf, wp[1]);

        v8f acc = (v8f)(0.0f);
        acc = wmma_bf16(ahi, bhi, acc);
        acc = wmma_bf16(ahi, blo, acc);
        acc = wmma_bf16(alo, bhi, acc);

        float s = 0.f, q = 0.f;
#pragma unroll
        for (int r = 0; r < 8; ++r) {
            int m = 8*half + r;
            y0[(size_t)(row0 + m) * 64 + n] = acc[r];
            s += acc[r]; q += acc[r] * acc[r];
        }
        s += __shfl_xor(s, 16);
        q += __shfl_xor(q, 16);
        if (half == 0) { atomicAdd(&bnSum[n], s); atomicAdd(&bnSq[n], q); }
    }
}

// --------------------------------------------------- BN param computation
__global__ void bnparams_kernel(const float* __restrict__ sum,
                                const float* __restrict__ sq,
                                const float* __restrict__ g,
                                const float* __restrict__ beta,
                                float* __restrict__ scale,
                                float* __restrict__ shift, int C) {
    int c = threadIdx.x;
    if (c >= C) return;
    float m   = sum[c] * (1.0f / NPTS);
    float var = sq[c]  * (1.0f / NPTS) - m * m;
    float s   = g[c] * rsqrtf(var + 1e-3f);
    scale[c] = s;
    shift[c] = beta[c] - m * s;
}

// ------------------------------ BN+ReLU in place + max-scatter into canvas
__global__ void bnrelu_scatter0_kernel(float* __restrict__ y0,
                                       const int* __restrict__ lin,
                                       const float* __restrict__ scale,
                                       const float* __restrict__ shift,
                                       float* __restrict__ v0) {
    long idx = (long)blockIdx.x * blockDim.x + threadIdx.x;
    if (idx >= (long)NPTS * 64) return;
    int i = (int)(idx >> 6), c = (int)(idx & 63);
    float v = fmaxf(fmaf(y0[idx], scale[c], shift[c]), 0.f);
    y0[idx] = v;   // y0 becomes p0
    atomicMax((unsigned int*)(v0 + (size_t)lin[i] * 64 + c), __float_as_uint(v));
}

// --------------- GEMM1: [p0 | v0[lin]] (N x 128) * W1 (128 x 128), 2 passes
template <bool FINAL>
__global__ __launch_bounds__(128)
void gemm1_kernel(const float* __restrict__ p0, const float* __restrict__ v0,
                  const int* __restrict__ lin, const unsigned* __restrict__ wpack,
                  const float* __restrict__ scale, const float* __restrict__ shift,
                  float* __restrict__ bnSum, float* __restrict__ bnSq,
                  float* __restrict__ out) {
    int tid  = threadIdx.x;
    int wave = tid >> 5, lane = tid & 31;
    int half = lane >> 4, mrow = lane & 15, ncol = lane & 15;
    int row0 = (blockIdx.x * 4 + wave) * 16;

    int arow = row0 + mrow;
    const float* pA = p0 + (size_t)arow * 64;
    const float* pV = v0 + (size_t)lin[arow] * 64;

    v8f acc[8];
#pragma unroll
    for (int nt = 0; nt < 8; ++nt) acc[nt] = (v8f)(0.0f);

#pragma unroll
    for (int ks = 0; ks < 4; ++ks) {
        int kb = ks * 32;
        v8u ah, al;
#pragma unroll
        for (int v = 0; v < 8; ++v) {
            int k = kb + 2*v + 16*half;
            float f0 = (k < 64) ? pA[k]   : pV[k - 64];
            float f1 = (k < 64) ? pA[k+1] : pV[k + 1 - 64];
            unsigned h, l;
            splitpack(f0, f1, h, l);
            ah[v] = h; al[v] = l;
        }
        v16bf ahi = __builtin_bit_cast(v16bf, ah);
        v16bf alo = __builtin_bit_cast(v16bf, al);

#pragma unroll
        for (int nt = 0; nt < 8; ++nt) {
            const v8u* wp = (const v8u*)(wpack + (size_t)(((ks*8 + nt)*32) + lane) * 16);
            v16bf bhi = __builtin_bit_cast(v16bf, wp[0]);
            v16bf blo = __builtin_bit_cast(v16bf, wp[1]);
            acc[nt] = wmma_bf16(ahi, bhi, acc[nt]);
            acc[nt] = wmma_bf16(ahi, blo, acc[nt]);
            acc[nt] = wmma_bf16(alo, bhi, acc[nt]);
        }
    }

    if (!FINAL) {
#pragma unroll
        for (int nt = 0; nt < 8; ++nt) {
            int n = nt*16 + ncol;
            float s = 0.f, q = 0.f;
#pragma unroll
            for (int r = 0; r < 8; ++r) { s += acc[nt][r]; q += acc[nt][r]*acc[nt][r]; }
            s += __shfl_xor(s, 16);
            q += __shfl_xor(q, 16);
            if (half == 0) { atomicAdd(&bnSum[n], s); atomicAdd(&bnSq[n], q); }
        }
    } else {
        int linr[8];
#pragma unroll
        for (int r = 0; r < 8; ++r) linr[r] = lin[row0 + 8*half + r];
#pragma unroll
        for (int nt = 0; nt < 8; ++nt) {
            int n = nt*16 + ncol;
            float sc = scale[n], sh = shift[n];
#pragma unroll
            for (int r = 0; r < 8; ++r) {
                float p = fmaxf(fmaf(acc[nt][r], sc, sh), 0.f);
                atomicMax((unsigned int*)(out + (size_t)linr[r] * 128 + n),
                          __float_as_uint(p));
            }
        }
    }
}

// ---------------------------------------------------------------- launcher
extern "C" void kernel_launch(void* const* d_in, const int* in_sizes, int n_in,
                              void* d_out, int out_size, void* d_ws, size_t ws_size,
                              hipStream_t stream) {
    (void)in_sizes; (void)n_in; (void)out_size; (void)ws_size;
    const float* features = (const float*)d_in[0];
    const int*   coors    = (const int*)d_in[1];
    const float* W0       = (const float*)d_in[2];
    const float* g0       = (const float*)d_in[3];
    const float* beta0    = (const float*)d_in[4];
    const float* W1       = (const float*)d_in[5];
    const float* g1       = (const float*)d_in[6];
    const float* beta1    = (const float*)d_in[7];
    float* out = (float*)d_out;

    char* ws = (char*)d_ws;
    size_t off = 0;
    auto alloc = [&](size_t bytes) {
        void* p = ws + off;
        off = (off + bytes + 255) & ~(size_t)255;
        return p;
    };
    int*      lin    = (int*)     alloc((size_t)NPTS * 4);
    float*    stats  = (float*)   alloc((size_t)CANVAS_SZ * 16);     // count + sum xyz
    float*    bn     = (float*)   alloc(4096);                       // BN accum/params
    float*    v0     = (float*)   alloc((size_t)CANVAS_SZ * 64 * 4); // layer-0 voxel max
    float*    x22    = (float*)   alloc((size_t)NPTS * 32 * 4);      // padded features
    float*    y0     = (float*)   alloc((size_t)NPTS * 64 * 4);      // y0 -> p0 in place
    unsigned* wpack0 = (unsigned*)alloc(1 * 4 * 32 * 16 * 4);        // 8 KB
    unsigned* wpack1 = (unsigned*)alloc(4 * 8 * 32 * 16 * 4);        // 64 KB

    float* bnSum0 = bn + 0,   *bnSq0 = bn + 64;
    float* bnSum1 = bn + 128, *bnSq1 = bn + 256;
    float* scale0 = bn + 384, *shift0 = bn + 448;
    float* scale1 = bn + 512, *shift1 = bn + 640;

    // zero stats + bn + v0 (contiguous region) and the output canvas
    long zero4 = ((long)CANVAS_SZ * 16 + 4096 + (long)CANVAS_SZ * 64 * 4) / 16;
    zero_kernel<<<2048, 256, 0, stream>>>((float4*)stats, zero4);
    zero_kernel<<<2048, 256, 0, stream>>>((float4*)out, (long)CANVAS_SZ * 128 / 4);

    // pre-pack weights into WMMA B-fragment hi/lo bf16 layout
    packW_kernel<<<1, 128, 0, stream>>>(W0, wpack0, 64, 4, 1, 22);
    packW_kernel<<<4, 256, 0, stream>>>(W1, wpack1, 128, 8, 4, 128);

    int ptBlocks = (NPTS + 255) / 256;
    point_stats_kernel<<<ptBlocks, 256, 0, stream>>>(features, coors, lin, stats);
    build_x22_kernel  <<<ptBlocks, 256, 0, stream>>>(features, coors, lin, stats, x22);

    int gemmBlocks = NPTS / 64;   // 4 waves * 16 rows per block
    gemm0_kernel<<<gemmBlocks, 128, 0, stream>>>(x22, wpack0, y0, bnSum0, bnSq0);
    bnparams_kernel<<<1, 128, 0, stream>>>(bnSum0, bnSq0, g0, beta0, scale0, shift0, 64);

    long el0 = (long)NPTS * 64;
    bnrelu_scatter0_kernel<<<(unsigned)((el0 + 255) / 256), 256, 0, stream>>>(
        y0, lin, scale0, shift0, v0);

    gemm1_kernel<false><<<gemmBlocks, 128, 0, stream>>>(
        y0, v0, lin, wpack1, scale1, shift1, bnSum1, bnSq1, out);
    bnparams_kernel<<<1, 128, 0, stream>>>(bnSum1, bnSq1, g1, beta1, scale1, shift1, 128);
    gemm1_kernel<true><<<gemmBlocks, 128, 0, stream>>>(
        y0, v0, lin, wpack1, scale1, shift1, bnSum1, bnSq1, out);
}
